// MixtureOfMixers_75007308857795
// MI455X (gfx1250) — matrix operand order repaired and checked by reading, not compiled, not run on hardware
//
#include <hip/hip_runtime.h>
#include <hip/hip_bf16.h>

// ---------------------------------------------------------------------------
// MixtureOfMixers on MI455X (gfx1250, wave32, WMMA)
// Every GEMM runs as v_wmma_f32_16x16x32_f16: each wave owns a 32x64 C tile
// (2x4 fragments). Per K=32 step: 6 fragment loads issued as one clause,
// then 8 back-to-back WMMAs. f16 operands / f32 accumulate; WGP-scope
// global_prefetch for the A/B streams.
// ---------------------------------------------------------------------------

typedef _Float16 v16h __attribute__((ext_vector_type(16)));
typedef _Float16 v8h  __attribute__((ext_vector_type(8)));
typedef float    v8f  __attribute__((ext_vector_type(8)));

#define Bb  32
#define Nn  512
#define Dd  1024
#define Hh  8
#define Ee  8
#define Kk  2
#define HD  128
#define HID 128

// ---- WMMA fragment helpers (wave32 layouts per CDNA5 ISA 7.12.2) ----------
// A (16x32 f16): lanes 0-15 row M=l hold K 0..7 (elems0-7) & 16..23 (elems8-15);
//                lanes 16-31 row M=l-16 hold K 8..15 & 24..31.
__device__ __forceinline__ v16h load_a_frag(const _Float16* __restrict__ A, int lda) {
  const int lane = threadIdx.x & 31;
  const int m  = lane & 15;
  const int k0 = (lane >> 4) << 3;          // 0 or 8
  const _Float16* p = A + (size_t)m * lda + k0;
  v8h lo = *(const v8h*)(p);                // K = k0 .. k0+7
  v8h hi = *(const v8h*)(p + 16);           // K = k0+16 .. k0+23
  v16h r;
#pragma unroll
  for (int i = 0; i < 8; ++i) { r[i] = lo[i]; r[i + 8] = hi[i]; }
  return r;
}

// B (32x16 f16), source stored as Bsrc[n][k] (K contiguous per column):
// lanes 0-15 col n=l hold K 0..7 & 16..23; lanes 16-31 col n=l-16 hold K 8..15 & 24..31.
__device__ __forceinline__ v16h load_b_frag(const _Float16* __restrict__ Bp, int ldb) {
  const int lane = threadIdx.x & 31;
  const int n  = lane & 15;
  const int k0 = (lane >> 4) << 3;          // 0 or 8
  const _Float16* p = Bp + (size_t)n * ldb + k0;
  v8h lo = *(const v8h*)(p);
  v8h hi = *(const v8h*)(p + 16);
  v16h r;
#pragma unroll
  for (int i = 0; i < 8; ++i) { r[i] = lo[i]; r[i + 8] = hi[i]; }
  return r;
}

__device__ __forceinline__ v8f wmma_f16(v16h a, v16h b, v8f c) {
  return __builtin_amdgcn_wmma_f32_16x16x32_f16(false, a, false, b, (short)0, c,
                                                false, false);
}

// One wave: 32x64 C tile = 2x4 fragments; A rows K-contig, Bsrc[n][k] K-contig.
// All 6 fragment loads are hoisted ahead of the 8 WMMAs so the scheduler can
// emit one load clause + single wait + a straight run of v_wmma ops.
__device__ __forceinline__ void mm32x64(const _Float16* __restrict__ A, int lda,
                                        const _Float16* __restrict__ Bp, int ldb,
                                        int K, v8f c[2][4]) {
  const int lane = threadIdx.x & 31;
  for (int k0 = 0; k0 < K; k0 += 32) {
    // speculative prefetch ~4 K-tiles ahead, WGP scope (locality 3);
    // lanes spread across the tile rows so each line is touched once
    __builtin_prefetch(A + (size_t)lane * lda + k0 + 128, 0, 3);
    __builtin_prefetch(Bp + (size_t)(lane * 2) * ldb + k0 + 128, 0, 3);
    v16h a0 = load_a_frag(A + k0, lda);
    v16h a1 = load_a_frag(A + (size_t)16 * lda + k0, lda);
    v16h b0 = load_b_frag(Bp + k0, ldb);
    v16h b1 = load_b_frag(Bp + (size_t)16 * ldb + k0, ldb);
    v16h b2 = load_b_frag(Bp + (size_t)32 * ldb + k0, ldb);
    v16h b3 = load_b_frag(Bp + (size_t)48 * ldb + k0, ldb);
    c[0][0] = wmma_f16(a0, b0, c[0][0]);
    c[1][0] = wmma_f16(a1, b0, c[1][0]);
    c[0][1] = wmma_f16(a0, b1, c[0][1]);
    c[1][1] = wmma_f16(a1, b1, c[1][1]);
    c[0][2] = wmma_f16(a0, b2, c[0][2]);
    c[1][2] = wmma_f16(a1, b2, c[1][2]);
    c[0][3] = wmma_f16(a0, b3, c[0][3]);
    c[1][3] = wmma_f16(a1, b3, c[1][3]);
  }
}

__device__ __forceinline__ void zero_acc(v8f c[2][4]) {
  v8f z = 0.0f;
#pragma unroll
  for (int i = 0; i < 2; ++i)
#pragma unroll
    for (int j = 0; j < 4; ++j) c[i][j] = z;
}

// C fragment element (i=0..7): row = rbase+i, col = coll  (wave32 layout)
#define C_EPILOGUE_VARS()                      \
  const int lane  = threadIdx.x & 31;          \
  const int coll  = lane & 15;                 \
  const int rbase = (lane >> 4) << 3;

// wave mapping inside a 64x128 block tile (4 waves, each 32x64)
#define WAVE_TILE_VARS()                                   \
  const int w  = threadIdx.x >> 5;                         \
  const int m0 = blockIdx.y * 64 + (w >> 1) * 32;          \
  const int n0 = blockIdx.x * 128 + (w & 1) * 64;

__device__ __forceinline__ float gelu_tanh(float x) {
  return 0.5f * x * (1.0f + tanhf(0.7978845608028654f * (x + 0.044715f * x * x * x)));
}

// ---------------------------- small kernels --------------------------------
__global__ void k_cvt_f16(const float* __restrict__ in, _Float16* __restrict__ out, int n) {
  int i = blockIdx.x * blockDim.x + threadIdx.x;
  if (i < n) out[i] = (_Float16)in[i];
}

__global__ void k_zero_f32(float* __restrict__ p, int n) {
  int i = blockIdx.x * blockDim.x + threadIdx.x;
  if (i < n) p[i] = 0.0f;
}

// token-mean: m[b][d] = mean_n x[b][n][d]
__global__ void k_mean(const float* __restrict__ x, float* __restrict__ m) {
  int i = blockIdx.x * blockDim.x + threadIdx.x;    // over Bb*Dd
  int b = i >> 10, d = i & 1023;
  const float* p = x + (size_t)b * Nn * Dd + d;
  float s = 0.0f;
  for (int n = 0; n < Nn; ++n) s += p[(size_t)n * Dd];
  m[i] = s * (1.0f / (float)Nn);
}

// per-batch router: logits, softmax, top-2, normalized weights (1 wave / batch)
__global__ void k_router(const float* __restrict__ m, const float* __restrict__ rw,
                         float* __restrict__ probs, int* __restrict__ ti,
                         float* __restrict__ tw) {
  int b = blockIdx.x;
  int lane = threadIdx.x;
  float logit[Ee];
#pragma unroll
  for (int e = 0; e < Ee; ++e) {
    float s = 0.0f;
    for (int d = lane; d < Dd; d += 32) s += m[b * Dd + d] * rw[e * Dd + d];
#pragma unroll
    for (int off = 16; off > 0; off >>= 1) s += __shfl_xor(s, off, 32);
    logit[e] = s;
  }
  if (lane == 0) {
    float mx = logit[0];
#pragma unroll
    for (int e = 1; e < Ee; ++e) mx = fmaxf(mx, logit[e]);
    float pe[Ee], sum = 0.0f;
#pragma unroll
    for (int e = 0; e < Ee; ++e) { pe[e] = expf(logit[e] - mx); sum += pe[e]; }
#pragma unroll
    for (int e = 0; e < Ee; ++e) { pe[e] /= sum; probs[b * Ee + e] = pe[e]; }
    int i0 = 0;
#pragma unroll
    for (int e = 1; e < Ee; ++e) if (pe[e] > pe[i0]) i0 = e;
    int i1 = (i0 == 0) ? 1 : 0;
#pragma unroll
    for (int e = 0; e < Ee; ++e) if (e != i0 && pe[e] > pe[i1]) i1 = e;
    float s2 = pe[i0] + pe[i1];
    ti[b * Kk + 0] = i0; ti[b * Kk + 1] = i1;
    tw[b * Kk + 0] = pe[i0] / s2; tw[b * Kk + 1] = pe[i1] / s2;
  }
}

// aux = E * sum_e mean_b(probs)[e] * mean_b(onehot(ti[:,0]))[e]
__global__ void k_aux(const float* __restrict__ probs, const int* __restrict__ ti,
                      float* __restrict__ out_aux) {
  int lane = threadIdx.x;
  float v = 0.0f;
  if (lane < Ee) {
    float ps = 0.0f; int cnt = 0;
    for (int b = 0; b < Bb; ++b) {
      ps += probs[b * Ee + lane];
      if (ti[b * Kk] == lane) cnt++;
    }
    v = (ps / (float)Bb) * ((float)cnt / (float)Bb);
  }
#pragma unroll
  for (int off = 16; off > 0; off >>= 1) v += __shfl_xor(v, off, 32);
  if (lane == 0) *out_aux = (float)Ee * v;
}

// ------------------------------ GEMM kernels -------------------------------
// xp = x @ in_w.T + in_b; store f16 TRANSPOSED as xt[b][dch][n]  (n contig)
__global__ void k_gemm_xp(const _Float16* __restrict__ xh, const _Float16* __restrict__ wh,
                          const float* __restrict__ bias, _Float16* __restrict__ xt) {
  WAVE_TILE_VARS();
  v8f c[2][4]; zero_acc(c);
  mm32x64(xh + (size_t)m0 * Dd, Dd, wh + (size_t)n0 * Dd, Dd, Dd, c);
  C_EPILOGUE_VARS();
#pragma unroll
  for (int si = 0; si < 2; ++si)
#pragma unroll
    for (int sj = 0; sj < 4; ++sj)
#pragma unroll
      for (int i = 0; i < 8; ++i) {
        int row = m0 + si * 16 + rbase + i;       // token = b*Nn + n
        int col = n0 + sj * 16 + coll;            // channel
        float v = c[si][sj][i] + bias[col];
        int b = row >> 9, n = row & 511;
        xt[((size_t)b * Dd + col) * Nn + n] = (_Float16)v;
      }
}

// LayerNorm over the token axis: one wave per (b, channel) row of 512
__global__ void k_layernorm(_Float16* __restrict__ xt) {
  int row  = blockIdx.x * (blockDim.x >> 5) + (threadIdx.x >> 5);  // Bb*Dd rows
  int lane = threadIdx.x & 31;
  _Float16* p = xt + (size_t)row * Nn;
  float s = 0.0f, s2 = 0.0f;
  for (int j = lane; j < Nn; j += 32) { float v = (float)p[j]; s += v; s2 += v * v; }
#pragma unroll
  for (int off = 16; off > 0; off >>= 1) { s += __shfl_xor(s, off, 32); s2 += __shfl_xor(s2, off, 32); }
  float mean = s * (1.0f / (float)Nn);
  float var  = s2 * (1.0f / (float)Nn) - mean * mean;
  float r = rsqrtf(var + 1e-5f);
  for (int j = lane; j < Nn; j += 32) {
    float v = (float)p[j];
    p[j] = (_Float16)((v - mean) * r);
  }
}

// h1a[d][e] = sum_n xn[b,h,d,n] * w1t[eid,h,e,n]; store TRANSPOSED h1aT[e][d]
__global__ void k_h1a(const _Float16* __restrict__ xt, const _Float16* __restrict__ w1t,
                      const int* __restrict__ ti, _Float16* __restrict__ h1aT) {
  int bhk = blockIdx.z;
  int b = bhk >> 4, h = (bhk >> 1) & 7, k = bhk & 1;
  int eid = ti[b * Kk + k];
  WAVE_TILE_VARS();
  v8f c[2][4]; zero_acc(c);
  const _Float16* A  = xt  + ((size_t)b * Dd + h * HD + m0) * Nn;
  const _Float16* Bp = w1t + (((size_t)eid * Hh + h) * HID + n0) * Nn;
  mm32x64(A, Nn, Bp, Nn, Nn, c);
  C_EPILOGUE_VARS();
  _Float16* out = h1aT + (size_t)bhk * HID * HD;
#pragma unroll
  for (int si = 0; si < 2; ++si)
#pragma unroll
    for (int sj = 0; sj < 4; ++sj)
#pragma unroll
      for (int i = 0; i < 8; ++i) {
        int row = m0 + si * 16 + rbase + i;   // d
        int col = n0 + sj * 16 + coll;        // e
        out[(size_t)col * HD + row] = (_Float16)c[si][sj][i];   // transposed
      }
}

// h1[c][e] = gelu( sum_d w1c[eid,h,c,d] * h1aT[e][d] + b1[eid,h,c,e] )
__global__ void k_h1(const _Float16* __restrict__ w1c, const _Float16* __restrict__ h1aT,
                     const float* __restrict__ b1, const int* __restrict__ ti,
                     _Float16* __restrict__ h1) {
  int bhk = blockIdx.z;
  int b = bhk >> 4, h = (bhk >> 1) & 7, k = bhk & 1;
  int eid = ti[b * Kk + k];
  WAVE_TILE_VARS();
  v8f c[2][4]; zero_acc(c);
  const _Float16* A  = w1c + ((size_t)eid * Hh + h) * HD * HD + (size_t)m0 * HD;
  const _Float16* Bp = h1aT + (size_t)bhk * HID * HD + (size_t)n0 * HD;
  mm32x64(A, HD, Bp, HD, HD, c);
  C_EPILOGUE_VARS();
  _Float16* out = h1 + (size_t)bhk * HD * HID;
  const float* bias = b1 + ((size_t)eid * Hh + h) * HD * HID;
#pragma unroll
  for (int si = 0; si < 2; ++si)
#pragma unroll
    for (int sj = 0; sj < 4; ++sj)
#pragma unroll
      for (int i = 0; i < 8; ++i) {
        int row = m0 + si * 16 + rbase + i;
        int col = n0 + sj * 16 + coll;
        float v = c[si][sj][i] + bias[(size_t)row * HID + col];
        out[(size_t)row * HID + col] = (_Float16)gelu_tanh(v);
      }
}

// o[dd][n] = sum_e h1[dd][e] * w2t[eid,h,n,e]; store TRANSPOSED oT[n][dd]
__global__ void k_o(const _Float16* __restrict__ h1, const _Float16* __restrict__ w2t,
                    const int* __restrict__ ti, _Float16* __restrict__ oT) {
  int bhk = blockIdx.z;
  int b = bhk >> 4, h = (bhk >> 1) & 7, k = bhk & 1;
  int eid = ti[b * Kk + k];
  WAVE_TILE_VARS();
  v8f c[2][4]; zero_acc(c);
  const _Float16* A  = h1 + (size_t)bhk * HD * HID + (size_t)m0 * HID;
  const _Float16* Bp = w2t + (((size_t)eid * Hh + h) * Nn + n0) * HID;
  mm32x64(A, HID, Bp, HID, HID, c);
  C_EPILOGUE_VARS();
  _Float16* out = oT + (size_t)bhk * Nn * HD;
#pragma unroll
  for (int si = 0; si < 2; ++si)
#pragma unroll
    for (int sj = 0; sj < 4; ++sj)
#pragma unroll
      for (int i = 0; i < 8; ++i) {
        int row = m0 + si * 16 + rbase + i;   // dd
        int col = n0 + sj * 16 + coll;        // n
        out[(size_t)col * HD + row] = (_Float16)c[si][sj][i];   // transposed
      }
}

// out2[c][n] = sum_d w2c[eid,h,c,d] * oT[n][d] + b2; accG += tw[k]*out2
__global__ void k_out2(const _Float16* __restrict__ w2c, const _Float16* __restrict__ oT,
                       const float* __restrict__ b2, const int* __restrict__ ti,
                       const float* __restrict__ tw, float* __restrict__ accG) {
  int bhk = blockIdx.z;
  int b = bhk >> 4, h = (bhk >> 1) & 7, k = bhk & 1;
  int eid = ti[b * Kk + k];
  float tk = tw[b * Kk + k];
  WAVE_TILE_VARS();
  v8f c[2][4]; zero_acc(c);
  const _Float16* A  = w2c + ((size_t)eid * Hh + h) * HD * HD + (size_t)m0 * HD;
  const _Float16* Bp = oT + (size_t)bhk * Nn * HD + (size_t)n0 * HD;
  mm32x64(A, HD, Bp, HD, HD, c);
  C_EPILOGUE_VARS();
  const float* bias = b2 + ((size_t)eid * Hh + h) * HD * Nn;
  float* acc = accG + ((size_t)b * Hh + h) * HD * Nn;
#pragma unroll
  for (int si = 0; si < 2; ++si)
#pragma unroll
    for (int sj = 0; sj < 4; ++sj)
#pragma unroll
      for (int i = 0; i < 8; ++i) {
        int row = m0 + si * 16 + rbase + i;   // c
        int col = n0 + sj * 16 + coll;        // n
        float v = c[si][sj][i] + bias[(size_t)row * Nn + col];
        atomicAdd(&acc[(size_t)row * Nn + col], tk * v);  // exactly 2 addends
      }
}

// accG[b,h,c,n] (f32) -> yh[b*Nn+n][h*HD+c] (f16, K-contig for final GEMM)
__global__ void k_acc2yh(const float* __restrict__ accG, _Float16* __restrict__ yh) {
  int i = blockIdx.x * blockDim.x + threadIdx.x;    // Bb*Hh*HD*Nn
  int n = i & 511, c = (i >> 9) & 127, h = (i >> 16) & 7, b = i >> 19;
  yh[((size_t)b * Nn + n) * Dd + h * HD + c] = (_Float16)accG[i];
}

// y = yh @ out_w.T + out_b  -> d_out f32 (B,N,D) row-major
__global__ void k_gemm_final(const _Float16* __restrict__ yh, const _Float16* __restrict__ wh,
                             const float* __restrict__ bias, float* __restrict__ out) {
  WAVE_TILE_VARS();
  v8f c[2][4]; zero_acc(c);
  mm32x64(yh + (size_t)m0 * Dd, Dd, wh + (size_t)n0 * Dd, Dd, Dd, c);
  C_EPILOGUE_VARS();
#pragma unroll
  for (int si = 0; si < 2; ++si)
#pragma unroll
    for (int sj = 0; sj < 4; ++sj)
#pragma unroll
      for (int i = 0; i < 8; ++i) {
        int row = m0 + si * 16 + rbase + i;
        int col = n0 + sj * 16 + coll;
        out[(size_t)row * Dd + col] = c[si][sj][i] + bias[col];
      }
}

// ------------------------------- launcher ----------------------------------
extern "C" void kernel_launch(void* const* d_in, const int* in_sizes, int n_in,
                              void* d_out, int out_size, void* d_ws, size_t ws_size,
                              hipStream_t stream) {
  const float* x        = (const float*)d_in[0];
  const float* router_w = (const float*)d_in[1];
  const float* in_w     = (const float*)d_in[2];
  const float* in_b     = (const float*)d_in[3];
  const float* out_w    = (const float*)d_in[4];
  const float* out_b    = (const float*)d_in[5];
  const float* w1t      = (const float*)d_in[6];
  const float* w1c      = (const float*)d_in[7];
  const float* b1       = (const float*)d_in[8];
  const float* w2t      = (const float*)d_in[9];
  const float* w2c      = (const float*)d_in[10];
  const float* b2       = (const float*)d_in[11];
  float* out = (float*)d_out;
  float* out_aux = out + (size_t)out_size - 1;

  unsigned char* wsp = (unsigned char*)d_ws;
  auto alloc = [&](size_t bytes) -> void* {
    void* p = (void*)wsp;
    wsp += (bytes + 255) & ~(size_t)255;
    return p;
  };
  const size_t TOK = (size_t)Bb * Nn;                 // 16384 tokens
  _Float16* xh     = (_Float16*)alloc(TOK * Dd * 2);              // x f16
  _Float16* xt     = (_Float16*)alloc((size_t)Bb * Dd * Nn * 2);  // xp^T / xn
  _Float16* in_wh  = (_Float16*)alloc((size_t)Dd * Dd * 2);
  _Float16* out_wh = (_Float16*)alloc((size_t)Dd * Dd * 2);
  _Float16* w1th   = (_Float16*)alloc((size_t)Ee * Hh * HID * Nn * 2);
  _Float16* w2th   = (_Float16*)alloc((size_t)Ee * Hh * Nn * HID * 2);
  _Float16* w1ch   = (_Float16*)alloc((size_t)Ee * Hh * HD * HD * 2);
  _Float16* w2ch   = (_Float16*)alloc((size_t)Ee * Hh * HD * HD * 2);
  float*    mbuf   = (float*)alloc((size_t)Bb * Dd * 4);
  float*    probs  = (float*)alloc((size_t)Bb * Ee * 4);
  int*      ti     = (int*)alloc((size_t)Bb * Kk * 4);
  float*    tw     = (float*)alloc((size_t)Bb * Kk * 4);
  _Float16* h1aT   = (_Float16*)alloc((size_t)Bb * Hh * Kk * HID * HD * 2);
  _Float16* h1     = (_Float16*)alloc((size_t)Bb * Hh * Kk * HD * HID * 2);
  _Float16* oT     = (_Float16*)alloc((size_t)Bb * Hh * Kk * Nn * HD * 2);
  float*    accG   = (float*)alloc((size_t)Bb * Hh * HD * Nn * 4);
  _Float16* yh     = (_Float16*)alloc(TOK * Dd * 2);
  (void)ws_size; (void)n_in; (void)in_sizes;

  // f32 -> f16 conversions
  k_cvt_f16<<<(int)((TOK * Dd + 255) / 256), 256, 0, stream>>>(x, xh, (int)(TOK * Dd));
  k_cvt_f16<<<(Dd * Dd + 255) / 256, 256, 0, stream>>>(in_w, in_wh, Dd * Dd);
  k_cvt_f16<<<(Dd * Dd + 255) / 256, 256, 0, stream>>>(out_w, out_wh, Dd * Dd);
  k_cvt_f16<<<(Ee * Hh * HID * Nn + 255) / 256, 256, 0, stream>>>(w1t, w1th, Ee * Hh * HID * Nn);
  k_cvt_f16<<<(Ee * Hh * Nn * HID + 255) / 256, 256, 0, stream>>>(w2t, w2th, Ee * Hh * Nn * HID);
  k_cvt_f16<<<(Ee * Hh * HD * HD + 255) / 256, 256, 0, stream>>>(w1c, w1ch, Ee * Hh * HD * HD);
  k_cvt_f16<<<(Ee * Hh * HD * HD + 255) / 256, 256, 0, stream>>>(w2c, w2ch, Ee * Hh * HD * HD);

  // router path + aux
  k_mean<<<(Bb * Dd) / 256, 256, 0, stream>>>(x, mbuf);
  k_router<<<Bb, 32, 0, stream>>>(mbuf, router_w, probs, ti, tw);
  k_aux<<<1, 32, 0, stream>>>(probs, ti, out_aux);

  // zero the top-k accumulator
  k_zero_f32<<<(Bb * Hh * HD * Nn) / 256, 256, 0, stream>>>(accG, Bb * Hh * HD * Nn);

  // xp = x @ in_w.T + b (store transposed [b][d][n]), then LayerNorm over n
  k_gemm_xp<<<dim3(Dd / 128, (int)(TOK / 64)), 128, 0, stream>>>(xh, in_wh, in_b, xt);
  k_layernorm<<<(Bb * Dd) / 8, 256, 0, stream>>>(xt);

  // expert GEMM chain, one (b,h,k) problem per blockIdx.z
  k_h1a <<<dim3(HID / 128, HD / 64, Bb * Hh * Kk), 128, 0, stream>>>(xt, w1th, ti, h1aT);
  k_h1  <<<dim3(HID / 128, HD / 64, Bb * Hh * Kk), 128, 0, stream>>>(w1ch, h1aT, b1, ti, h1);
  k_o   <<<dim3(Nn / 128, HD / 64, Bb * Hh * Kk), 128, 0, stream>>>(h1, w2th, ti, oT);
  k_out2<<<dim3(Nn / 128, HD / 64, Bb * Hh * Kk), 128, 0, stream>>>(w2ch, oT, b2, ti, tw, accG);

  // combine -> final projection
  k_acc2yh<<<(Bb * Hh * HD * Nn) / 256, 256, 0, stream>>>(accG, yh);
  k_gemm_final<<<dim3(Dd / 128, (int)(TOK / 64)), 128, 0, stream>>>(yh, out_wh, out_b, out);
}
